// LSTM__78228534329353
// MI455X (gfx1250) — matrix operand order repaired and checked by reading, not compiled, not run on hardware
//
#include <hip/hip_runtime.h>
#include <hip/hip_bf16.h>

typedef __bf16 bf16_t;
typedef __bf16 v16bf __attribute__((ext_vector_type(16)));
typedef __bf16 v8bf  __attribute__((ext_vector_type(8)));
typedef float  v8f   __attribute__((ext_vector_type(8)));
typedef float  v4f   __attribute__((ext_vector_type(4)));

#define LB 64      // batch
#define LS 1024    // seq len
#define LI 256     // input dim
#define LH 512     // hidden dim
#define LG 2048    // 4*LH
#define LK 768     // LI + LH

#define NBLK 32          // one block per hidden 16-col tile
#define NTHR 128         // 4 waves: one batch 16-row tile each

// LDS row strides (bf16 elements), padded to break bank alignment
#define WPAD 784   // 768 + 16
#define HPAD 528   // 512 + 16
#define XPAD 272   // 256 + 16
#define HLDS_BYTE_BASE (64 * WPAD * 2)                       // hlds offset in LDS
#define SMEM_BYTES ((64 * WPAD + 64 * HPAD + 64 * XPAD) * 2) // 202,752 B < 320 KB

// ---------------------------------------------------------------------------
// WMMA operand loaders (CDNA5 ISA layouts):
//  A (16-bit 16x32): lanes 0-15: row=lane, K {0..7,16..23}; lanes 16-31: K {8..15,24..31}
//  B (16-bit 32x16): lanes 0-15: col=lane, K {0..7,16..23}; lanes 16-31: K {8..15,24..31}
// ---------------------------------------------------------------------------
__device__ __forceinline__ v16bf load_a_lds(const bf16_t* row, int k0, int khalf) {
    v8bf c0 = *(const v8bf*)(row + k0 + khalf * 8);
    v8bf c1 = *(const v8bf*)(row + k0 + 16 + khalf * 8);
    v16bf a;
#pragma unroll
    for (int i = 0; i < 8; ++i) { a[i] = c0[i]; a[i + 8] = c1[i]; }
    return a;
}
__device__ __forceinline__ v16bf load_b_lds(const bf16_t* wrow, int k0, int khalf) {
    v8bf c0 = *(const v8bf*)(wrow + k0 + khalf * 8);
    v8bf c1 = *(const v8bf*)(wrow + k0 + 16 + khalf * 8);
    v16bf b;
#pragma unroll
    for (int i = 0; i < 8; ++i) { b[i] = c0[i]; b[i + 8] = c1[i]; }
    return b;
}
__device__ __forceinline__ float sigmoidf_(float v) {
    return 1.0f / (1.0f + __expf(-v));
}

struct Acc4 { v8f a0, a1, a2, a3; };

// x-side partial gates: K = 0..255 (independent of the recurrence)
__device__ __forceinline__ Acc4 compute_xside(const bf16_t* arx,
        const bf16_t* w0, const bf16_t* w1, const bf16_t* w2, const bf16_t* w3,
        int khalf) {
    Acc4 r; r.a0 = {}; r.a1 = {}; r.a2 = {}; r.a3 = {};
#pragma unroll 2
    for (int k0 = 0; k0 < LI; k0 += 32) {
        v16bf a  = load_a_lds(arx, k0, khalf);
        v16bf b0 = load_b_lds(w0, k0, khalf);
        v16bf b1 = load_b_lds(w1, k0, khalf);
        v16bf b2 = load_b_lds(w2, k0, khalf);
        v16bf b3 = load_b_lds(w3, k0, khalf);
        r.a0 = __builtin_amdgcn_wmma_f32_16x16x32_bf16(false, a, false, b0, (short)0, r.a0, false, false);
        r.a1 = __builtin_amdgcn_wmma_f32_16x16x32_bf16(false, a, false, b1, (short)0, r.a1, false, false);
        r.a2 = __builtin_amdgcn_wmma_f32_16x16x32_bf16(false, a, false, b2, (short)0, r.a2, false, false);
        r.a3 = __builtin_amdgcn_wmma_f32_16x16x32_bf16(false, a, false, b3, (short)0, r.a3, false, false);
    }
    return r;
}
// h-side accumulation: K = 256..767 (the serial part)
__device__ __forceinline__ void accum_hside(Acc4& r, const bf16_t* arh,
        const bf16_t* w0, const bf16_t* w1, const bf16_t* w2, const bf16_t* w3,
        int khalf) {
#pragma unroll 2
    for (int kh = 0; kh < LH; kh += 32) {
        v16bf a  = load_a_lds(arh, kh, khalf);
        v16bf b0 = load_b_lds(w0, LI + kh, khalf);
        v16bf b1 = load_b_lds(w1, LI + kh, khalf);
        v16bf b2 = load_b_lds(w2, LI + kh, khalf);
        v16bf b3 = load_b_lds(w3, LI + kh, khalf);
        r.a0 = __builtin_amdgcn_wmma_f32_16x16x32_bf16(false, a, false, b0, (short)0, r.a0, false, false);
        r.a1 = __builtin_amdgcn_wmma_f32_16x16x32_bf16(false, a, false, b1, (short)0, r.a1, false, false);
        r.a2 = __builtin_amdgcn_wmma_f32_16x16x32_bf16(false, a, false, b2, (short)0, r.a2, false, false);
        r.a3 = __builtin_amdgcn_wmma_f32_16x16x32_bf16(false, a, false, b3, (short)0, r.a3, false, false);
    }
}

// ---------------------------------------------------------------------------
// Prep: pack weights -> Wcat[2048][768] bf16 (gates i,f,g,o; cols 0..255 input
// side, 256..767 hidden side), bias = b_i + b_h, zero h0 and sync counters.
// ---------------------------------------------------------------------------
__global__ void lstm_prep(
    const float* __restrict__ W_ii, const float* __restrict__ b_ii,
    const float* __restrict__ W_hi, const float* __restrict__ b_hi,
    const float* __restrict__ W_if, const float* __restrict__ b_if,
    const float* __restrict__ W_hf, const float* __restrict__ b_hf,
    const float* __restrict__ W_ig, const float* __restrict__ b_ig,
    const float* __restrict__ W_hg, const float* __restrict__ b_hg,
    const float* __restrict__ W_io, const float* __restrict__ b_io,
    const float* __restrict__ W_ho, const float* __restrict__ b_ho,
    bf16_t* __restrict__ Wcat, float* __restrict__ bias,
    bf16_t* __restrict__ h0, int* __restrict__ syncv)
{
    int idx = blockIdx.x * blockDim.x + threadIdx.x;
    int stride = gridDim.x * blockDim.x;

    for (int i = idx; i < LG * LK; i += stride) {
        int r = i / LK;
        int col = i - r * LK;
        int g = r >> 9;
        int u = r & (LH - 1);
        const float* Wi = (g == 0) ? W_ii : (g == 1) ? W_if : (g == 2) ? W_ig : W_io;
        const float* Wh = (g == 0) ? W_hi : (g == 1) ? W_hf : (g == 2) ? W_hg : W_ho;
        float v = (col < LI) ? Wi[u * LI + col] : Wh[u * LH + (col - LI)];
        Wcat[i] = (bf16_t)v;
    }
    for (int i = idx; i < LG; i += stride) {
        int g = i >> 9;
        int u = i & (LH - 1);
        const float* bi = (g == 0) ? b_ii : (g == 1) ? b_if : (g == 2) ? b_ig : b_io;
        const float* bh = (g == 0) ? b_hi : (g == 1) ? b_hf : (g == 2) ? b_hg : b_ho;
        bias[i] = bi[u] + bh[u];
    }
    for (int i = idx; i < LB * LH; i += stride) h0[i] = (bf16_t)0.0f;
    if (idx < 2) syncv[idx] = 0;
}

// ---------------------------------------------------------------------------
// Device-wide barrier (32 resident blocks). Fences make h visible across WGPs.
// ---------------------------------------------------------------------------
__device__ __forceinline__ void grid_sync(int* count, int* gen) {
    __threadfence();          // flush this thread's h stores to device scope
    __syncthreads();
    if (threadIdx.x == 0) {
        int g = __hip_atomic_load(gen, __ATOMIC_ACQUIRE, __HIP_MEMORY_SCOPE_AGENT);
        int arrived = __hip_atomic_fetch_add(count, 1, __ATOMIC_ACQ_REL, __HIP_MEMORY_SCOPE_AGENT);
        if (arrived == NBLK - 1) {
            __hip_atomic_store(count, 0, __ATOMIC_RELEASE, __HIP_MEMORY_SCOPE_AGENT);
            __hip_atomic_fetch_add(gen, 1, __ATOMIC_RELEASE, __HIP_MEMORY_SCOPE_AGENT);
        } else {
            while (__hip_atomic_load(gen, __ATOMIC_ACQUIRE, __HIP_MEMORY_SCOPE_AGENT) == g)
                __builtin_amdgcn_s_sleep(1);
        }
    }
    __syncthreads();
    __threadfence();          // acquire: invalidate stale near-cache lines
}

// ---------------------------------------------------------------------------
// Persistent LSTM: 32 blocks x 128 threads. Block owns hidden cols
// [16*nt, 16*nt+16); wave owns batch rows [16*mt, 16*mt+16). Weights staged in
// LDS once; h staged per step via async global->LDS; x staged one step ahead
// and its partial gates computed outside the serial path; c stays in VGPRs.
// ---------------------------------------------------------------------------
__global__ void __launch_bounds__(NTHR)
lstm_persist(const float* __restrict__ x,
             const bf16_t* __restrict__ Wcat,
             const float* __restrict__ bias,
             bf16_t* __restrict__ hbuf0,
             bf16_t* __restrict__ hbuf1,
             float* __restrict__ out,
             float* __restrict__ h_last,
             float* __restrict__ c_last,
             int* __restrict__ sync_count,
             int* __restrict__ sync_gen)
{
    extern __shared__ char smem[];
    bf16_t* wlds = (bf16_t*)smem;            // 64 rows x WPAD
    bf16_t* hlds = wlds + 64 * WPAD;         // 64 rows x HPAD (byte base HLDS_BYTE_BASE)
    bf16_t* xlds = hlds + 64 * HPAD;         // 64 rows x XPAD

    const int tid   = threadIdx.x;
    const int lane  = tid & 31;
    const int wave  = tid >> 5;              // 0..3 = batch tile mt
    const int nt    = blockIdx.x;            // 0..31 hidden tile
    const int mbase = wave * 16;
    const int nbase = nt * 16;
    const int lrow  = lane & 15;
    const int khalf = lane >> 4;

    // ---- Stage this block's weight slice into LDS (once, reused 1024x) ----
    for (int ci = tid; ci < 64 * 96; ci += NTHR) {           // 16B chunks
        int row = ci / 96;
        int cc  = ci - row * 96;
        int gate = row >> 4;
        int r    = row & 15;
        const uint4* src = (const uint4*)(Wcat + (size_t)(gate * LH + nbase + r) * LK + cc * 8);
        uint4* dst = (uint4*)(wlds + row * WPAD + cc * 8);
        *dst = *src;
    }
    // ---- Stage x_0 (f32 -> bf16) into LDS ----
    for (int ci = tid; ci < 64 * 32; ci += NTHR) {           // 8-float chunks
        int row = ci >> 5;
        int cc  = ci & 31;
        const v4f* src = (const v4f*)(x + (size_t)row * (LS * LI) + cc * 8);
        v4f f0 = src[0], f1 = src[1];
        v8bf o;
#pragma unroll
        for (int i = 0; i < 4; ++i) { o[i] = (bf16_t)f0[i]; o[i + 4] = (bf16_t)f1[i]; }
        *(v8bf*)(xlds + row * XPAD + cc * 8) = o;
    }
    __syncthreads();

    // Per-wave constants held in registers.
    const int col = nbase + lrow;
    const float bsi = bias[0 * LH + col];
    const float bsf = bias[1 * LH + col];
    const float bsg = bias[2 * LH + col];
    const float bso = bias[3 * LH + col];

    const bf16_t* w0 = wlds + (0 * 16 + lrow) * WPAD;
    const bf16_t* w1 = wlds + (1 * 16 + lrow) * WPAD;
    const bf16_t* w2 = wlds + (2 * 16 + lrow) * WPAD;
    const bf16_t* w3 = wlds + (3 * 16 + lrow) * WPAD;
    const bf16_t* arx = xlds + (mbase + lrow) * XPAD;
    const bf16_t* arh = hlds + (mbase + lrow) * HPAD;
    const int mrow0 = mbase + khalf * 8;

    Acc4 xacc = compute_xside(arx, w0, w1, w2, w3, khalf);   // x partial for t=0
    v8f cacc = {};   // cell state, register-resident for the whole sequence

    bf16_t* hb[2] = {hbuf0, hbuf1};

    for (int t = 0; t < LS; ++t) {
        const bf16_t* h_in  = hb[t & 1];
        bf16_t*       h_out = hb[(t + 1) & 1];

        // ---- Stage h (bf16, 64x512) via async global->LDS (ASYNCcnt) ----
        for (int ci = tid; ci < 64 * 64; ci += NTHR) {       // 16B chunks/lane
            int row = ci >> 6;
            int cc  = ci & 63;
            unsigned lds_off = (unsigned)(HLDS_BYTE_BASE + row * (HPAD * 2) + cc * 16);
            unsigned long long gaddr = (unsigned long long)(size_t)(h_in + row * LH + cc * 8);
            asm volatile("global_load_async_to_lds_b128 %0, %1, off"
                         :: "v"(lds_off), "v"(gaddr) : "memory");
        }
        asm volatile("s_wait_asynccnt 0x0" ::: "memory");
        __syncthreads();

        // ---- Serial part: h-side K accumulation on top of x partials ----
        Acc4 acc = xacc;
        accum_hside(acc, arh, w0, w1, w2, w3, khalf);

        // ---- Fused gate math + state update (c in registers) ----
#pragma unroll
        for (int v = 0; v < 8; ++v) {
            int m = mrow0 + v;
            float ig = sigmoidf_(acc.a0[v] + bsi);
            float fg = sigmoidf_(acc.a1[v] + bsf);
            float gg = tanhf(acc.a2[v] + bsg);
            float og = sigmoidf_(acc.a3[v] + bso);
            float c1 = fg * cacc[v] + ig * gg;
            float h1 = og * tanhf(c1);
            cacc[v] = c1;
            out[(size_t)m * (LS * LH) + (size_t)t * LH + col] = h1;
            h_out[m * LH + col] = (bf16_t)h1;
            if (t == LS - 1) {
                h_last[m * LH + col] = h1;
                c_last[m * LH + col] = c1;
            }
        }

        // ---- Off the serial path: stage x_{t+1}, compute its partial gates,
        //      and prefetch x_{t+2}; overlaps with laggard blocks + barrier ----
        if (t + 1 < LS) {
            __syncthreads();    // xlds readers of step t+1's partials are done below
            for (int ci = tid; ci < 64 * 32; ci += NTHR) {
                int row = ci >> 5;
                int cc  = ci & 31;
                const v4f* src = (const v4f*)(x + (size_t)row * (LS * LI) + (size_t)(t + 1) * LI + cc * 8);
                v4f f0 = src[0], f1 = src[1];
                v8bf o;
#pragma unroll
                for (int i = 0; i < 4; ++i) { o[i] = (bf16_t)f0[i]; o[i + 4] = (bf16_t)f1[i]; }
                *(v8bf*)(xlds + row * XPAD + cc * 8) = o;
            }
            if (t + 2 < LS) {
                for (int j = tid; j < 512; j += NTHR) {
                    int row = j >> 3;
                    int off = (j & 7) * 32;
                    __builtin_prefetch(x + (size_t)row * (LS * LI) + (size_t)(t + 2) * LI + off, 0, 0);
                }
            }
            __syncthreads();
            xacc = compute_xside(arx, w0, w1, w2, w3, khalf);
        }

        grid_sync(sync_count, sync_gen);
    }
}

extern "C" void kernel_launch(void* const* d_in, const int* in_sizes, int n_in,
                              void* d_out, int out_size, void* d_ws, size_t ws_size,
                              hipStream_t stream)
{
    (void)in_sizes; (void)n_in; (void)out_size; (void)ws_size;

    const float* x    = (const float*)d_in[0];
    const float* W_ii = (const float*)d_in[1];
    const float* b_ii = (const float*)d_in[2];
    const float* W_hi = (const float*)d_in[3];
    const float* b_hi = (const float*)d_in[4];
    const float* W_if = (const float*)d_in[5];
    const float* b_if = (const float*)d_in[6];
    const float* W_hf = (const float*)d_in[7];
    const float* b_hf = (const float*)d_in[8];
    const float* W_ig = (const float*)d_in[9];
    const float* b_ig = (const float*)d_in[10];
    const float* W_hg = (const float*)d_in[11];
    const float* b_hg = (const float*)d_in[12];
    const float* W_io = (const float*)d_in[13];
    const float* b_io = (const float*)d_in[14];
    const float* W_ho = (const float*)d_in[15];
    const float* b_ho = (const float*)d_in[16];

    char* ws = (char*)d_ws;
    bf16_t* Wcat  = (bf16_t*)ws; ws += (size_t)LG * LK * sizeof(bf16_t);   // 3 MB
    float*  bias  = (float*)ws;  ws += (size_t)LG * sizeof(float);
    bf16_t* hbuf0 = (bf16_t*)ws; ws += (size_t)LB * LH * sizeof(bf16_t);
    bf16_t* hbuf1 = (bf16_t*)ws; ws += (size_t)LB * LH * sizeof(bf16_t);
    int*    syncv = (int*)ws;    ws += 256;

    float* out    = (float*)d_out;
    float* h_last = out + (size_t)LB * LS * LH;
    float* c_last = h_last + (size_t)LB * LH;

    lstm_prep<<<512, 256, 0, stream>>>(W_ii, b_ii, W_hi, b_hi, W_if, b_if,
                                       W_hf, b_hf, W_ig, b_ig, W_hg, b_hg,
                                       W_io, b_io, W_ho, b_ho,
                                       Wcat, bias, hbuf0, syncv);

    lstm_persist<<<NBLK, NTHR, SMEM_BYTES, stream>>>(x, Wcat, bias,
                                                     hbuf0, hbuf1,
                                                     out, h_last, c_last,
                                                     syncv, syncv + 1);
}
